// ADMBlock_3685081940031
// MI455X (gfx1250) — matrix operand order repaired and checked by reading, not compiled, not run on hardware
//
#include <hip/hip_runtime.h>
#include <math.h>

// ---------------------------------------------------------------- constants
constexpr int N_    = 4096;
constexpr int D_    = 256;
constexpr int P_    = 64;
constexpr int H_    = 8;
constexpr int DH_   = 32;
constexpr int GH_   = 4;
constexpr int F_    = 64;
constexpr int K_    = 32;
constexpr int A_    = 5;
constexpr int NBINS_= 16;
constexpr int NREL_ = 66;
constexpr int NC_   = 8;
constexpr int NB_   = 2;

// ---------------------------------------------------------------- WMMA types
typedef __attribute__((ext_vector_type(16))) __bf16 v16bf;
typedef __attribute__((ext_vector_type(8)))  float  v8f;

__device__ __forceinline__ float gelu_tanh(float x) {
  float x3 = x * x * x;
  return 0.5f * x * (1.0f + tanhf(0.7978845608028654f * (x + 0.044715f * x3)));
}

// ---------------------------------------------------------------- B pre-pack: f32 weights -> WMMA bf16 lane layout
// chunk index t = ((ktile*Nd + ncol)*2 + half); each chunk = 16 bf16 (K = ktile*32 + 8*half + e for e<8, ktile*32 + 16 + 8*half + (e-8) for e>=8)
__global__ void pack_b_kernel(const float* __restrict__ B, int ldb, int Kd, int Nd,
                              __bf16* __restrict__ Bp) {
  int t = blockIdx.x * blockDim.x + threadIdx.x;
  int total = (Kd / 32) * Nd * 2;
  if (t >= total) return;
  int half  = t & 1;
  int rest  = t >> 1;
  int ncol  = rest % Nd;
  int ktile = rest / Nd;
  const float* src = B + (size_t)(ktile * 32 + 8 * half) * ldb + ncol;
  __bf16* dst = Bp + (size_t)t * 16;
  #pragma unroll
  for (int e = 0; e < 16; ++e) dst[e] = (__bf16)src[(size_t)(e + (e & 8)) * ldb];
}

// ---------------------------------------------------------------- generic WMMA GEMM (pre-packed B)
// C[M,Nd] = act( A[M,KD] @ B[KD,Nd] + bias ) (+ C if accum)
// One wave per 16x16 C tile. KD compile-time multiple of 32; M,Nd multiples of 16.
template <int KD>
__global__ void gemm_wmma_kernel(const float* __restrict__ A, int lda,
                                 const __bf16* __restrict__ Bp, int Nd,
                                 float* __restrict__ C, int ldc,
                                 const float* __restrict__ bias,
                                 int act, int accum) {
  const int mt    = blockIdx.x;
  const int nt    = blockIdx.y;
  const int lane  = threadIdx.x;      // 0..31
  const int lhalf = (lane >> 4) & 1;  // 0 or 1
  const int lmod  = lane & 15;

  v8f c = {0.f,0.f,0.f,0.f,0.f,0.f,0.f,0.f};

  const float* Arow = A + (size_t)(mt * 16 + lmod) * lda;
  const int ncol = nt * 16 + lmod;
  const v16bf* Bv = reinterpret_cast<const v16bf*>(Bp);

  #pragma unroll 4
  for (int kk = 0; kk < KD; kk += 32) {
    __builtin_prefetch(Arow + kk + 32, 0, 0);  // global_prefetch_b8
    // A 16x32 bf16 layout: lane = M (mod 16);
    // needed K indices = {8*lhalf + 0..7} and {16 + 8*lhalf + 0..7} -> two float4 pairs
    const float4* A4a = reinterpret_cast<const float4*>(Arow + kk + 8 * lhalf);
    const float4* A4b = reinterpret_cast<const float4*>(Arow + kk + 16 + 8 * lhalf);
    float4 a0 = A4a[0], a1 = A4a[1];
    float4 a2 = A4b[0], a3 = A4b[1];
    v16bf a;
    a[0]=(__bf16)a0.x; a[1]=(__bf16)a0.y; a[2]=(__bf16)a0.z; a[3]=(__bf16)a0.w;
    a[4]=(__bf16)a1.x; a[5]=(__bf16)a1.y; a[6]=(__bf16)a1.z; a[7]=(__bf16)a1.w;
    a[8]=(__bf16)a2.x; a[9]=(__bf16)a2.y; a[10]=(__bf16)a2.z; a[11]=(__bf16)a2.w;
    a[12]=(__bf16)a3.x; a[13]=(__bf16)a3.y; a[14]=(__bf16)a3.z; a[15]=(__bf16)a3.w;
    // B: one 32-byte contiguous chunk per lane, already in register layout
    v16bf b = Bv[(size_t)((kk >> 5) * Nd + ncol) * 2 + lhalf];
    c = __builtin_amdgcn_wmma_f32_16x16x32_bf16(false, a, false, b,
                                                (short)0, c, false, false);
  }

  #pragma unroll
  for (int r = 0; r < 8; ++r) {
    int m = mt * 16 + r + (lhalf ? 8 : 0);
    int n = nt * 16 + lmod;
    float val = c[r];
    if (bias) val += bias[n];
    if (act == 1) val = gelu_tanh(val);
    size_t off = (size_t)m * ldc + n;
    if (accum) val += C[off];
    C[off] = val;
  }
}

// ---------------------------------------------------------------- layer norm (row per block, 64 thr)
__global__ void layernorm_kernel(const float* __restrict__ x,
                                 const float* __restrict__ g,
                                 const float* __restrict__ b,
                                 float* __restrict__ y, int cols) {
  __shared__ float red[64];
  const int row = blockIdx.x;
  const float* xr = x + (size_t)row * cols;
  float s = 0.f;
  for (int i = threadIdx.x; i < cols; i += 64) s += xr[i];
  red[threadIdx.x] = s; __syncthreads();
  for (int st = 32; st > 0; st >>= 1) {
    if ((int)threadIdx.x < st) red[threadIdx.x] += red[threadIdx.x + st];
    __syncthreads();
  }
  float mean = red[0] / cols; __syncthreads();
  float v = 0.f;
  for (int i = threadIdx.x; i < cols; i += 64) { float d = xr[i] - mean; v += d * d; }
  red[threadIdx.x] = v; __syncthreads();
  for (int st = 32; st > 0; st >>= 1) {
    if ((int)threadIdx.x < st) red[threadIdx.x] += red[threadIdx.x + st];
    __syncthreads();
  }
  float rstd = rsqrtf(red[0] / cols + 1e-5f);
  float* yr = y + (size_t)row * cols;
  for (int i = threadIdx.x; i < cols; i += 64) yr[i] = (xr[i] - mean) * rstd * g[i] + b[i];
}

// ---------------------------------------------------------------- pair init: rp1[onehot] + fiP[n] + fjP[nb]
__global__ void pair_init_kernel(const int* __restrict__ nb, const int* __restrict__ resi,
                                 const int* __restrict__ chain, const int* __restrict__ batch,
                                 const float* __restrict__ rp, const float* __restrict__ fiP,
                                 const float* __restrict__ fjP, float* __restrict__ pair) {
  int t = blockIdx.x * blockDim.x + threadIdx.x;
  if (t >= N_ * K_) return;
  int n = t / K_;
  int j = nb[t];
  int js = (j < 0) ? 0 : j;
  int d = resi[js] - resi[n];
  d = (d < -32 ? -32 : (d > 32 ? 32 : d)) + 32;
  bool same = (j >= 0) && (chain[js] == chain[n]) && (batch[js] == batch[n]);
  int idx = same ? d : (NREL_ - 1);
  const float* rpr = rp + (size_t)idx * P_;
  const float* fir = fiP + (size_t)n * P_;
  const float* fjr = fjP + (size_t)js * P_;
  float* pr = pair + (size_t)t * P_;
  #pragma unroll 8
  for (int p = 0; p < P_; ++p) pr[p] = rpr[p] + fir[p] + fjr[p];
}

// ---------------------------------------------------------------- pair @ wb -> logits bias (N,K,H)
__global__ void pair_bias_kernel(const float* __restrict__ pair, const float* __restrict__ wb,
                                 float* __restrict__ blog) {
  int t = blockIdx.x * blockDim.x + threadIdx.x;
  if (t >= N_ * K_) return;
  const float* pr = pair + (size_t)t * P_;
  float acc[H_];
  #pragma unroll
  for (int h = 0; h < H_; ++h) acc[h] = 0.f;
  for (int p = 0; p < P_; ++p) {
    float pv = pr[p];
    const float* w = wb + (size_t)p * H_;
    #pragma unroll
    for (int h = 0; h < H_; ++h) acc[h] += pv * w[h];
  }
  float* o = blog + (size_t)t * H_;
  #pragma unroll
  for (int h = 0; h < H_; ++h) o[h] = acc[h];
}

// ---------------------------------------------------------------- attention logits + softmax (wave per (n,h), lane=k)
__global__ void attn_softmax_kernel(const float* __restrict__ q, const float* __restrict__ kbuf,
                                    const float* __restrict__ blog, const int* __restrict__ nb,
                                    const unsigned char* __restrict__ mask,
                                    const unsigned char* __restrict__ pos_mask, int use_posmask,
                                    float* __restrict__ attn) {
  int wid = (blockIdx.x * blockDim.x + threadIdx.x) >> 5;
  int lane = threadIdx.x & 31;
  if (wid >= N_ * H_) return;
  int n = wid / H_, h = wid % H_;
  int j = nb[n * K_ + lane];
  int js = (j < 0) ? 0 : j;
  const float* qr = q + (size_t)n * (H_ * DH_) + h * DH_;
  const float* kr = kbuf + (size_t)js * (H_ * DH_) + h * DH_;
  float s = 0.f;
  #pragma unroll
  for (int d = 0; d < DH_; ++d) s += qr[d] * kr[d];
  s *= 0.17677669529663687f;                       // 1/sqrt(32)
  s += blog[(size_t)(n * K_ + lane) * H_ + h];
  bool pm = mask[n] && (j >= 0) && mask[js];
  if (use_posmask) pm = pm && (pos_mask[n] != 0);
  if (!pm) s = -1e9f;
  float m = s;
  #pragma unroll
  for (int o = 16; o > 0; o >>= 1) m = fmaxf(m, __shfl_xor(m, o, 32));
  float e = expf(s - m);
  float sum = e;
  #pragma unroll
  for (int o = 16; o > 0; o >>= 1) sum += __shfl_xor(sum, o, 32);
  attn[(size_t)wid * K_ + lane] = e / sum;
}

// ---------------------------------------------------------------- sum_k attn * v[nb]  -> (N,256)
__global__ void attn_out_v_kernel(const float* __restrict__ attn, const float* __restrict__ vbuf,
                                  const int* __restrict__ nb, float* __restrict__ out) {
  int t = blockIdx.x * blockDim.x + threadIdx.x;
  if (t >= N_ * H_ * DH_) return;
  int n = t >> 8; int c = t & 255; int h = c >> 5;
  const float* ar = attn + ((size_t)n * H_ + h) * K_;
  const int* nbr = nb + n * K_;
  float acc = 0.f;
  #pragma unroll 4
  for (int k = 0; k < K_; ++k) {
    int j = nbr[k]; int js = (j < 0) ? 0 : j;
    acc += ar[k] * vbuf[(size_t)js * 256 + c];
  }
  out[t] = acc;
}

// ---------------------------------------------------------------- sum_k attn * pair -> (N,H,P)
__global__ void attn_wpair_kernel(const float* __restrict__ attn, const float* __restrict__ pair,
                                  float* __restrict__ wpair) {
  int t = blockIdx.x * blockDim.x + threadIdx.x;
  if (t >= N_ * H_ * P_) return;
  int n = t / (H_ * P_); int rem = t % (H_ * P_); int h = rem / P_; int p = rem % P_;
  const float* ar = attn + ((size_t)n * H_ + h) * K_;
  const float* pr = pair + (size_t)n * K_ * P_ + p;
  float acc = 0.f;
  #pragma unroll 4
  for (int k = 0; k < K_; ++k) acc += ar[k] * pr[(size_t)k * P_];
  wpair[t] = acc;
}

// ---------------------------------------------------------------- local frames
__global__ void frames_kernel(const float* __restrict__ pos, float* __restrict__ Rm) {
  int n = blockIdx.x * blockDim.x + threadIdx.x;
  if (n >= N_) return;
  const float* pn = pos + (size_t)n * A_ * 3;
  float Nv[3], ca[3], Cv[3], e1[3], e2[3], e3[3];
  for (int i = 0; i < 3; ++i) { Nv[i] = pn[i]; ca[i] = pn[3 + i]; Cv[i] = pn[6 + i]; }
  float nr = 0.f;
  for (int i = 0; i < 3; ++i) { e1[i] = Cv[i] - ca[i]; nr += e1[i] * e1[i]; }
  nr = sqrtf(nr) + 1e-8f;
  for (int i = 0; i < 3; ++i) e1[i] /= nr;
  float v2[3], dot = 0.f;
  for (int i = 0; i < 3; ++i) { v2[i] = Nv[i] - ca[i]; dot += v2[i] * e1[i]; }
  nr = 0.f;
  for (int i = 0; i < 3; ++i) { e2[i] = v2[i] - dot * e1[i]; nr += e2[i] * e2[i]; }
  nr = sqrtf(nr) + 1e-8f;
  for (int i = 0; i < 3; ++i) e2[i] /= nr;
  e3[0] = e1[1] * e2[2] - e1[2] * e2[1];
  e3[1] = e1[2] * e2[0] - e1[0] * e2[2];
  e3[2] = e1[0] * e2[1] - e1[1] * e2[0];
  float* R = Rm + (size_t)n * 9;
  for (int i = 0; i < 3; ++i) { R[i * 3 + 0] = e1[i]; R[i * 3 + 1] = e2[i]; R[i * 3 + 2] = e3[i]; }
}

// ---------------------------------------------------------------- geometric pair2 build
__global__ void pair2_geo_kernel(const float* __restrict__ pos, const float* __restrict__ Rm,
                                 const int* __restrict__ nb, const int* __restrict__ resi,
                                 const int* __restrict__ chain, const int* __restrict__ batch,
                                 const float* __restrict__ rp2, const float* __restrict__ wdist,
                                 const float* __restrict__ wdir, const float* __restrict__ wrot,
                                 const float* __restrict__ wpvec, float* __restrict__ pair2) {
  int t = blockIdx.x * blockDim.x + threadIdx.x;
  if (t >= N_ * K_) return;
  int n = t / K_;
  int j = nb[t]; int js = (j < 0) ? 0 : j;
  int d = resi[js] - resi[n];
  d = (d < -32 ? -32 : (d > 32 ? 32 : d)) + 32;
  bool same = (j >= 0) && (chain[js] == chain[n]) && (batch[js] == batch[n]);
  int idx = same ? d : (NREL_ - 1);

  float can[3], caj[3];
  for (int i = 0; i < 3; ++i) { can[i] = pos[((size_t)n * A_ + 1) * 3 + i];
                                caj[i] = pos[((size_t)js * A_ + 1) * 3 + i]; }
  float dd = 0.f;
  for (int i = 0; i < 3; ++i) { float dx = caj[i] - can[i]; dd += dx * dx; }
  float dist = sqrtf(dd);

  float gbin[NBINS_];
  const float sigma = 22.0f / NBINS_;
  #pragma unroll
  for (int b = 0; b < NBINS_; ++b) {
    float c = 22.0f * (float)b / (float)(NBINS_ - 1);
    float df = dist - c;
    gbin[b] = expf(-df * df / (2.0f * sigma * sigma));
  }

  const float* R = Rm + (size_t)n * 9;
  float dirs[15], pvv[15], rot[9];
  for (int a = 0; a < A_; ++a) {
    float v[3], w[3], nr = 0.f;
    for (int i = 0; i < 3; ++i) {
      v[i] = caj[i] - pos[((size_t)n * A_ + a) * 3 + i];
      w[i] = pos[((size_t)js * A_ + a) * 3 + i] - can[i];
      nr += v[i] * v[i];
    }
    nr = sqrtf(nr) + 1e-8f;
    for (int i = 0; i < 3; ++i) v[i] /= nr;
    for (int i = 0; i < 3; ++i) {
      float dv = 0.f, pw = 0.f;
      for (int jj = 0; jj < 3; ++jj) { dv += R[jj * 3 + i] * v[jj]; pw += R[jj * 3 + i] * w[jj]; }
      dirs[a * 3 + i] = dv; pvv[a * 3 + i] = pw;
    }
  }
  const float* Rj = Rm + (size_t)js * 9;
  for (int i = 0; i < 3; ++i)
    for (int l = 0; l < 3; ++l) {
      float s = 0.f;
      for (int jj = 0; jj < 3; ++jj) s += R[jj * 3 + i] * Rj[jj * 3 + l];
      rot[i * 3 + l] = s;
    }

  float* pr = pair2 + (size_t)t * P_;
  for (int p = 0; p < P_; ++p) {
    float acc = rp2[(size_t)idx * P_ + p];
    #pragma unroll
    for (int b = 0; b < NBINS_; ++b) acc += gbin[b] * wdist[(size_t)b * P_ + p];
    #pragma unroll
    for (int u = 0; u < 15; ++u) acc += dirs[u] * wdir[(size_t)u * P_ + p];
    #pragma unroll
    for (int u = 0; u < 9; ++u) acc += rot[u] * wrot[(size_t)u * P_ + p];
    #pragma unroll
    for (int u = 0; u < 15; ++u) acc += pvv[u] * wpvec[(size_t)u * P_ + p];
    pr[p] = acc;
  }
}

// ---------------------------------------------------------------- zero rows where mask==0
__global__ void rowmask_kernel(float* __restrict__ buf, const unsigned char* __restrict__ m,
                               int rowlen, size_t total) {
  size_t t = (size_t)blockIdx.x * blockDim.x + threadIdx.x;
  if (t >= total) return;
  if (!m[t / rowlen]) buf[t] = 0.f;
}

// ---------------------------------------------------------------- segment counts (chain: 0..7, batch: 8..9)
__global__ void seg_counts_kernel(const int* __restrict__ chain, const int* __restrict__ batch,
                                  const unsigned char* __restrict__ mask, float* __restrict__ cnts) {
  int t = threadIdx.x;
  if (t < NC_) {
    float c = 0.f;
    for (int n = 0; n < N_; ++n) if (mask[n] && chain[n] == t) c += 1.f;
    cnts[t] = c;
  } else if (t < NC_ + NB_) {
    int b = t - NC_;
    float c = 0.f;
    for (int n = 0; n < N_; ++n) if (mask[n] && batch[n] == b) c += 1.f;
    cnts[t] = c;
  }
}

// ---------------------------------------------------------------- bias segment mean over batch (NB x 512)
__global__ void bias_segmean_kernel(const float* __restrict__ bias_pre, const int* __restrict__ batch,
                                    const unsigned char* __restrict__ mask,
                                    const float* __restrict__ cnts, float* __restrict__ segbias) {
  int t = blockIdx.x * blockDim.x + threadIdx.x;
  if (t >= NB_ * 512) return;
  int seg = t / 512, col = t % 512;
  float s = 0.f;
  for (int n = 0; n < N_; ++n)
    if (mask[n] && batch[n] == seg) s += bias_pre[(size_t)n * 512 + col];
  segbias[t] = s / fmaxf(cnts[NC_ + seg], 1e-6f);
}

// ---------------------------------------------------------------- kv outer-product segment means
__global__ void kv_segmean_kernel(const float* __restrict__ gk, const float* __restrict__ gv,
                                  const int* __restrict__ segarr, const unsigned char* __restrict__ mask,
                                  const float* __restrict__ cnts, int cntoff, int nseg,
                                  float* __restrict__ out) {
  int t = blockIdx.x * blockDim.x + threadIdx.x;
  const int per = GH_ * F_ * F_;
  int seg = t / per;
  if (seg >= nseg) return;
  int rem = t % per; int h = rem / (F_ * F_); int vk = rem % (F_ * F_);
  int vv = vk / F_; int kk = vk % F_;
  float acc = 0.f;
  for (int n = 0; n < N_; ++n)
    if (mask[n] && segarr[n] == seg)
      acc += gk[(size_t)n * 256 + h * F_ + vv] * gv[(size_t)n * 256 + h * F_ + kk];
  out[t] = acc / fmaxf(cnts[cntoff + seg], 1e-6f);
}

// ---------------------------------------------------------------- final gating einsum + bias -> (N,512)
__global__ void global_out_kernel(const float* __restrict__ chainmean, const float* __restrict__ batchmean,
                                  const float* __restrict__ gq, const float* __restrict__ segbias,
                                  const int* __restrict__ chain, const int* __restrict__ batch,
                                  float* __restrict__ outg) {
  int t = blockIdx.x * blockDim.x + threadIdx.x;
  if (t >= N_ * 512) return;
  int n = t >> 9; int col = t & 511; int h = col >> 7; int v = col & 127;
  const float* qr = gq + (size_t)n * 256 + h * F_;
  const float* mrow;
  if (v < F_) mrow = chainmean + (((size_t)chain[n] * GH_ + h) * F_ + v) * F_;
  else        mrow = batchmean + (((size_t)batch[n] * GH_ + h) * F_ + (v - F_)) * F_;
  float acc = 0.f;
  #pragma unroll 8
  for (int k = 0; k < F_; ++k) acc += mrow[k] * qr[k];
  acc += segbias[(size_t)batch[n] * 512 + col];
  outg[t] = acc;
}

// ================================================================ host side
static inline void gemm(const float* A, int lda, const float* B, int ldb,
                        float* C, int ldc, int M, int Kd, int Nd,
                        const float* bias, int act, int accum,
                        __bf16* bscratch, hipStream_t s) {
  int packn = (Kd / 32) * Nd * 2;
  pack_b_kernel<<<(packn + 255) / 256, 256, 0, s>>>(B, ldb, Kd, Nd, bscratch);
  dim3 g(M / 16, Nd / 16);
  switch (Kd) {
    case 64:  gemm_wmma_kernel<64> <<<g, 32, 0, s>>>(A, lda, bscratch, Nd, C, ldc, bias, act, accum); break;
    case 128: gemm_wmma_kernel<128><<<g, 32, 0, s>>>(A, lda, bscratch, Nd, C, ldc, bias, act, accum); break;
    case 256: gemm_wmma_kernel<256><<<g, 32, 0, s>>>(A, lda, bscratch, Nd, C, ldc, bias, act, accum); break;
    case 512: gemm_wmma_kernel<512><<<g, 32, 0, s>>>(A, lda, bscratch, Nd, C, ldc, bias, act, accum); break;
    default: break; // all GEMMs in this pipeline use Kd in {64,128,256,512}
  }
}

extern "C" void kernel_launch(void* const* d_in, const int* in_sizes, int n_in,
                              void* d_out, int out_size, void* d_ws, size_t ws_size,
                              hipStream_t stream) {
  (void)in_sizes; (void)n_in; (void)out_size; (void)ws_size;
  // ---------------- inputs
  const float* features = (const float*)d_in[0];
  const float* pos      = (const float*)d_in[1];
  const unsigned char* pos_mask = (const unsigned char*)d_in[2];
  const int* resi_nb = (const int*)d_in[3];
  const int* pos_nb  = (const int*)d_in[4];
  const int* resi    = (const int*)d_in[5];
  const int* chain   = (const int*)d_in[6];
  const int* batch   = (const int*)d_in[7];
  const unsigned char* mask = (const unsigned char*)d_in[8];
  // ---------------- params (setup_inputs dict order)
  const float* ln_f_g = (const float*)d_in[9];
  const float* ln_f_b = (const float*)d_in[10];
  const float* rp1    = (const float*)d_in[11];
  const float* fi     = (const float*)d_in[12];
  const float* fj     = (const float*)d_in[13];
  const float* ln_p1_g= (const float*)d_in[14];
  const float* ln_p1_b= (const float*)d_in[15];
  const float* mlp1_w1= (const float*)d_in[16];
  const float* mlp1_b1= (const float*)d_in[17];
  const float* mlp1_w2= (const float*)d_in[18];
  const float* mlp1_b2= (const float*)d_in[19];
  const float* ln1_g  = (const float*)d_in[20];
  const float* ln1_b  = (const float*)d_in[21];
  const float* wq1    = (const float*)d_in[22];
  const float* wk1    = (const float*)d_in[23];
  const float* wv1    = (const float*)d_in[24];
  const float* wb1    = (const float*)d_in[25];
  const float* wpv1   = (const float*)d_in[26];
  const float* wo1    = (const float*)d_in[27];
  const float* rp2    = (const float*)d_in[28];
  const float* wdist  = (const float*)d_in[29];
  const float* wdir   = (const float*)d_in[30];
  const float* wrot   = (const float*)d_in[31];
  const float* wpvec  = (const float*)d_in[32];
  const float* ln_p2_g= (const float*)d_in[33];
  const float* ln_p2_b= (const float*)d_in[34];
  const float* mlp2_w1= (const float*)d_in[35];
  const float* mlp2_b1= (const float*)d_in[36];
  const float* mlp2_w2= (const float*)d_in[37];
  const float* mlp2_b2= (const float*)d_in[38];
  const float* ln2_g  = (const float*)d_in[39];
  const float* ln2_b  = (const float*)d_in[40];
  const float* wq2    = (const float*)d_in[41];
  const float* wk2    = (const float*)d_in[42];
  const float* wv2    = (const float*)d_in[43];
  const float* wb2    = (const float*)d_in[44];
  const float* wpv2   = (const float*)d_in[45];
  const float* wo2    = (const float*)d_in[46];
  const float* wgk    = (const float*)d_in[47];
  const float* bgk    = (const float*)d_in[48];
  const float* wgv    = (const float*)d_in[49];
  const float* bgv    = (const float*)d_in[50];
  const float* wgq    = (const float*)d_in[51];
  const float* bgq    = (const float*)d_in[52];
  const float* wbias  = (const float*)d_in[53];
  const float* bbias  = (const float*)d_in[54];
  const float* wgo    = (const float*)d_in[55];

  // ---------------- workspace layout (floats)
  float* ws = (float*)d_ws;
  const size_t ND = (size_t)N_ * D_;
  float* xln   = ws;                        // N*D
  float* x1    = xln   + ND;                // N*D
  float* x2    = x1    + ND;                // N*D
  float* fiP   = x2    + ND;                // N*P
  float* fjP   = fiP   + (size_t)N_ * P_;   // N*P
  float* pair  = fjP   + (size_t)N_ * P_;   // N*K*P   (reused for pair2)
  float* hid   = pair  + (size_t)N_ * K_ * P_;       // N*K*2P
  float* qb    = hid   + (size_t)N_ * K_ * 2 * P_;   // N*256 (reused gk)
  float* kb    = qb    + ND;                // N*256 (reused gv)
  float* vb    = kb    + ND;                // N*256 (reused gq)
  float* blog  = vb    + ND;                // N*K*H
  float* attn  = blog  + (size_t)N_ * K_ * H_;       // N*H*K
  float* attno = attn  + (size_t)N_ * H_ * K_;       // N*256
  float* wpair = attno + ND;                // N*H*P
  float* Rm    = wpair + (size_t)N_ * H_ * P_;       // N*9 (padded to 16)
  float* b512  = Rm    + (size_t)N_ * 16;   // N*512 (bias_pre, reused as outg)
  float* segb  = b512  + (size_t)N_ * 512;  // NB*512
  float* cnts  = segb  + (size_t)NB_ * 512; // 16
  float* chS   = cnts  + 16;                // NC*GH*F*F
  float* baS   = chS   + (size_t)NC_ * GH_ * F_ * F_; // NB*GH*F*F
  __bf16* bpk  = (__bf16*)(baS + (size_t)NB_ * GH_ * F_ * F_); // 512*512 bf16 max

  const int NK = N_ * K_;
  float* out = (float*)d_out;

  // features running buffer = d_out
  hipMemcpyAsync(out, features, ND * sizeof(float), hipMemcpyDeviceToDevice, stream);

  // ---- pair branch 1
  layernorm_kernel<<<N_, 64, 0, stream>>>(features, ln_f_g, ln_f_b, xln, D_);
  layernorm_kernel<<<N_, 64, 0, stream>>>(features, ln1_g, ln1_b, x1, D_);
  gemm(xln, D_, fi, P_, fiP, P_, N_, D_, P_, nullptr, 0, 0, bpk, stream);
  gemm(xln, D_, fj, P_, fjP, P_, N_, D_, P_, nullptr, 0, 0, bpk, stream);
  pair_init_kernel<<<NK / 256, 256, 0, stream>>>(resi_nb, resi, chain, batch, rp1, fiP, fjP, pair);
  layernorm_kernel<<<NK, 64, 0, stream>>>(pair, ln_p1_g, ln_p1_b, pair, P_);
  gemm(pair, P_, mlp1_w1, 2 * P_, hid, 2 * P_, NK, P_, 2 * P_, mlp1_b1, 1, 0, bpk, stream);
  gemm(hid, 2 * P_, mlp1_w2, P_, pair, P_, NK, 2 * P_, P_, mlp1_b2, 0, 0, bpk, stream);

  // ---- sparse attention 1
  gemm(x1, D_, wq1, 256, qb, 256, N_, D_, 256, nullptr, 0, 0, bpk, stream);
  gemm(x1, D_, wk1, 256, kb, 256, N_, D_, 256, nullptr, 0, 0, bpk, stream);
  gemm(x1, D_, wv1, 256, vb, 256, N_, D_, 256, nullptr, 0, 0, bpk, stream);
  pair_bias_kernel<<<NK / 256, 256, 0, stream>>>(pair, wb1, blog);
  attn_softmax_kernel<<<(N_ * H_ * 32) / 256, 256, 0, stream>>>(qb, kb, blog, resi_nb, mask, pos_mask, 0, attn);
  attn_out_v_kernel<<<(N_ * 256) / 256, 256, 0, stream>>>(attn, vb, resi_nb, attno);
  attn_wpair_kernel<<<(N_ * H_ * P_) / 256, 256, 0, stream>>>(attn, pair, wpair);
  for (int h = 0; h < H_; ++h)
    gemm(wpair + h * P_, H_ * P_, wpv1 + h * DH_, 256, attno + h * DH_, 256,
         N_, P_, DH_, nullptr, 0, 1, bpk, stream);
  gemm(attno, 256, wo1, D_, out, D_, N_, 256, D_, nullptr, 0, 1, bpk, stream);  // features += ...

  // ---- geometric pair branch 2 (pair buffer reused)
  frames_kernel<<<N_ / 256, 256, 0, stream>>>(pos, Rm);
  pair2_geo_kernel<<<NK / 256, 256, 0, stream>>>(pos, Rm, pos_nb, resi, chain, batch,
                                                 rp2, wdist, wdir, wrot, wpvec, pair);
  layernorm_kernel<<<NK, 64, 0, stream>>>(pair, ln_p2_g, ln_p2_b, pair, P_);
  gemm(pair, P_, mlp2_w1, 2 * P_, hid, 2 * P_, NK, P_, 2 * P_, mlp2_b1, 1, 0, bpk, stream);
  gemm(hid, 2 * P_, mlp2_w2, P_, pair, P_, NK, 2 * P_, P_, mlp2_b2, 0, 0, bpk, stream);
  rowmask_kernel<<<(unsigned)(((size_t)NK * P_) / 256), 256, 0, stream>>>(pair, pos_mask, K_ * P_, (size_t)NK * P_);

  // ---- sparse attention 2
  layernorm_kernel<<<N_, 64, 0, stream>>>(out, ln2_g, ln2_b, x2, D_);
  gemm(x2, D_, wq2, 256, qb, 256, N_, D_, 256, nullptr, 0, 0, bpk, stream);
  gemm(x2, D_, wk2, 256, kb, 256, N_, D_, 256, nullptr, 0, 0, bpk, stream);
  gemm(x2, D_, wv2, 256, vb, 256, N_, D_, 256, nullptr, 0, 0, bpk, stream);
  pair_bias_kernel<<<NK / 256, 256, 0, stream>>>(pair, wb2, blog);
  attn_softmax_kernel<<<(N_ * H_ * 32) / 256, 256, 0, stream>>>(qb, kb, blog, pos_nb, mask, pos_mask, 1, attn);
  attn_out_v_kernel<<<(N_ * 256) / 256, 256, 0, stream>>>(attn, vb, pos_nb, attno);
  attn_wpair_kernel<<<(N_ * H_ * P_) / 256, 256, 0, stream>>>(attn, pair, wpair);
  for (int h = 0; h < H_; ++h)
    gemm(wpair + h * P_, H_ * P_, wpv2 + h * DH_, 256, attno + h * DH_, 256,
         N_, P_, DH_, nullptr, 0, 1, bpk, stream);
  rowmask_kernel<<<(unsigned)(ND / 256), 256, 0, stream>>>(attno, pos_mask, 256, ND);
  gemm(attno, 256, wo2, D_, out, D_, N_, 256, D_, nullptr, 0, 1, bpk, stream);  // features += where(pos_mask,...)

  // ---- global gating block (gk->qb, gv->kb, gq->vb)
  gemm(out, D_, wgk, 256, qb, 256, N_, D_, 256, bgk, 1, 0, bpk, stream);   // keyv = gelu(...)
  gemm(out, D_, wgv, 256, kb, 256, N_, D_, 256, bgv, 0, 0, bpk, stream);   // valv
  gemm(out, D_, wgq, 256, vb, 256, N_, D_, 256, bgq, 1, 0, bpk, stream);   // qv = gelu(...)
  gemm(out, D_, wbias, 512, b512, 512, N_, D_, 512, bbias, 0, 0, bpk, stream);  // bias_pre
  seg_counts_kernel<<<1, 64, 0, stream>>>(chain, batch, mask, cnts);
  bias_segmean_kernel<<<(NB_ * 512) / 256, 256, 0, stream>>>(b512, batch, mask, cnts, segb);
  kv_segmean_kernel<<<(NC_ * GH_ * F_ * F_) / 256, 256, 0, stream>>>(qb, kb, chain, mask, cnts, 0, NC_, chS);
  kv_segmean_kernel<<<(NB_ * GH_ * F_ * F_) / 256, 256, 0, stream>>>(qb, kb, batch, mask, cnts, NC_, NB_, baS);
  // b512 free now -> reuse as outg
  global_out_kernel<<<(N_ * 512) / 256, 256, 0, stream>>>(chS, baS, vb, segb, chain, batch, b512);
  gemm(b512, 512, wgo, D_, out, D_, N_, 512, D_, nullptr, 0, 1, bpk, stream);   // features += out @ wgo
}
